// RoPESelfAttention_74801150427529
// MI455X (gfx1250) — matrix operand order repaired and checked by reading, not compiled, not run on hardware
//
#include <hip/hip_runtime.h>
#include <math.h>
#include <stdint.h>

typedef __bf16 bf16_t;
typedef __attribute__((ext_vector_type(16))) __bf16 v16bf;
typedef __attribute__((ext_vector_type(8)))  __bf16 v8bf;
typedef __attribute__((ext_vector_type(8)))  float  v8f;

// ---------------------------------------------------------------------------
// helpers
// ---------------------------------------------------------------------------
__device__ inline v16bf load16(const bf16_t* p0, const bf16_t* p1) {
  v8bf lo = *reinterpret_cast<const v8bf*>(p0);
  v8bf hi = *reinterpret_cast<const v8bf*>(p1);
  v16bf r;
#pragma unroll
  for (int e = 0; e < 8; ++e) { r[e] = lo[e]; r[e + 8] = hi[e]; }
  return r;
}

__device__ inline v8f wmma_bf16(v16bf a, v16bf b, v8f c) {
  return __builtin_amdgcn_wmma_f32_16x16x32_bf16(
      /*neg_a=*/false, a, /*neg_b=*/false, b,
      /*c_mod=*/(short)0, c, /*reuse_a=*/false, /*reuse_b=*/false);
}

// Async DMA: one 16-byte chunk global -> LDS, tracked on ASYNCcnt.
// VDST = LDS byte offset (low 32 bits of generic LDS pointer), VADDR = 64b addr.
__device__ inline void async_ld_b128(const bf16_t* lds_ptr, const bf16_t* gptr) {
  unsigned lds_off = (unsigned)(uintptr_t)lds_ptr;
  unsigned long long ga = (unsigned long long)(uintptr_t)gptr;
  asm volatile("global_load_async_to_lds_b128 %0, %1, off"
               :: "v"(lds_off), "v"(ga)
               : "memory");
}

// ---------------------------------------------------------------------------
// fp32 -> bf16 convert
// ---------------------------------------------------------------------------
__global__ void f32_to_bf16_kernel(const float* __restrict__ in,
                                   bf16_t* __restrict__ out, size_t n) {
  size_t i = (size_t)blockIdx.x * blockDim.x + threadIdx.x;
  if (i < n) out[i] = (bf16_t)in[i];
}

// ---------------------------------------------------------------------------
// C[M,N] = A[M,K] * B[N,K]^T   (bf16 inputs, fp32 out)
// block = 128 threads = 4 waves (2x2), wave tile = 64x64 (16 WMMA C tiles)
// K loop split: main phase with unconditional WGP-scope prefetch, short tail.
// ---------------------------------------------------------------------------
__global__ __launch_bounds__(128) void gemm_bf16_wmma(
    const bf16_t* __restrict__ A, const bf16_t* __restrict__ Bm,
    float* __restrict__ C, int M, int N, int K) {
  const int lane = threadIdx.x & 31;
  const int wid  = threadIdx.x >> 5;
  const int lmod = lane & 15;
  const int lhi  = lane >> 4;
  const int m0 = blockIdx.y * 128 + (wid >> 1) * 64;
  const int n0 = blockIdx.x * 128 + (wid & 1) * 64;

  v8f c[4][4] = {};
  const int akb = lhi * 8;    // A: K = akb..akb+7 and 16+akb..16+akb+7
  const int bkb = lhi * 8;    // B: K = bkb..bkb+7 and 16+bkb..16+bkb+7

  auto kstep = [&](int k0, bool pf) __attribute__((always_inline)) {
    v16bf a[4], b[4];
#pragma unroll
    for (int i = 0; i < 4; ++i) {
      const bf16_t* pa = A + (size_t)(m0 + i * 16 + lmod) * K + k0 + akb;
      a[i] = load16(pa, pa + 16);
      if (pf) __builtin_prefetch(pa + 128, 0, 3);   // WGP-scope, 4 k-steps ahead
    }
#pragma unroll
    for (int j = 0; j < 4; ++j) {
      const bf16_t* pb = Bm + (size_t)(n0 + j * 16 + lmod) * K + k0 + bkb;
      b[j] = load16(pb, pb + 16);
      if (pf) __builtin_prefetch(pb + 128, 0, 3);
    }
#pragma unroll
    for (int i = 0; i < 4; ++i)
#pragma unroll
      for (int j = 0; j < 4; ++j)
        c[i][j] = wmma_bf16(a[i], b[j], c[i][j]);
  };

  int k0 = 0;
  for (; k0 + 128 < K; k0 += 32) kstep(k0, true);   // steady state, no branches
  for (; k0 < K; k0 += 32)       kstep(k0, false);  // tail (4 iters)

#pragma unroll
  for (int i = 0; i < 4; ++i)
#pragma unroll
    for (int j = 0; j < 4; ++j)
#pragma unroll
      for (int r = 0; r < 8; ++r) {
        int row = m0 + i * 16 + r + 8 * lhi;
        int col = n0 + j * 16 + lmod;
        C[(size_t)row * N + col] = c[i][j][r];
      }
}

// ---------------------------------------------------------------------------
// RoPE on q,k + split/convert to WMMA-friendly bf16 layouts:
//   qb, kb : (B*H, L, 64)   row-major
//   vt     : (B*H, 64, L)   (transposed so PV B-frags are contiguous)
// ---------------------------------------------------------------------------
__global__ void rope_split_kernel(const float* __restrict__ qkv,
                                  bf16_t* __restrict__ qb,
                                  bf16_t* __restrict__ kb,
                                  bf16_t* __restrict__ vt) {
  const int D = 1024, H = 16, L = 2048, TD = 3 * 1024;
  size_t idx = (size_t)blockIdx.x * blockDim.x + threadIdx.x;
  size_t total = (size_t)2 * L * TD;
  if (idx >= total) return;

  int e    = (int)(idx % TD);
  size_t bl = idx / TD;           // b*L + l
  int l = (int)(bl % L);
  int part = e / D;               // 0=q 1=k 2=v
  int cc = e % D;
  int h = cc >> 6;
  int d = cc & 63;
  size_t bh = (bl / L) * H + h;

  float val = qkv[idx];
  if (part < 2) {
    int dm = d & 31;
    float inv = __powf(10000.0f, -(float)dm * (1.0f / 32.0f));
    float ang = (float)l * inv;
    float sn, cs;
    __sincosf(ang, &sn, &cs);
    int od = (d < 32) ? d + 32 : d - 32;
    float other = qkv[bl * TD + (size_t)part * D + h * 64 + od];
    float out = (d < 32) ? (val * cs - other * sn) : (val * cs + other * sn);
    bf16_t* dst = (part == 0) ? qb : kb;
    dst[(bh * L + l) * 64 + d] = (bf16_t)out;
  } else {
    vt[(bh * 64 + d) * L + l] = (bf16_t)val;
  }
}

// ---------------------------------------------------------------------------
// Flash attention. One (b,h) per blockIdx.y, 64 query rows per block,
// 4 waves x 16 rows. KV tile = 64, double-buffered K/V staged into LDS with
// async DMA (ASYNCcnt), online softmax, P re-materialized as A-frag via LDS.
// out: bf16 (B, L, D) with col = h*64 + d
// ---------------------------------------------------------------------------
__global__ __launch_bounds__(128) void attn_kernel(
    const bf16_t* __restrict__ qb, const bf16_t* __restrict__ kb,
    const bf16_t* __restrict__ vt, bf16_t* __restrict__ out) {
  const int L = 2048, Dh = 64;
  __shared__ __align__(16) bf16_t sK[2][64 * 64];  // 2 x 8 KB, row = kv, col = dh
  __shared__ __align__(16) bf16_t sV[2][64 * 64];  // 2 x 8 KB, row = dh, col = kv
  __shared__ __align__(16) bf16_t sP[4][16 * 64];  // per-wave P tiles

  const int tid  = threadIdx.x;
  const int lane = tid & 31;
  const int wid  = tid >> 5;
  const int lmod = lane & 15;
  const int lhi  = lane >> 4;
  const int bh = blockIdx.y;
  const int q0 = blockIdx.x * 64 + wid * 16;

  const bf16_t* gK = kb + (size_t)bh * L * Dh;   // rows contiguous (64 bf16 each)
  const bf16_t* gV = vt + (size_t)bh * Dh * L;   // row (dh) stride L

  // Q A-fragments for Dh = 0..31 and 32..63
  const bf16_t* Q = qb + ((size_t)bh * L + q0) * Dh;
  v16bf aq[2];
#pragma unroll
  for (int s = 0; s < 2; ++s) {
    const bf16_t* p = Q + (size_t)lmod * Dh + s * 32 + lhi * 8;
    aq[s] = load16(p, p + 16);
  }

  float mrow[8], lrow[8];
#pragma unroll
  for (int r = 0; r < 8; ++r) { mrow[r] = -3.0e38f; lrow[r] = 0.0f; }
  v8f o[4] = {};
  const float scale = 0.125f; // 1/sqrt(64)

  // stage one 64-row K/V tile into LDS buffer `buf`: 8 async b128 per thread
  auto stage = [&](int kv0, int buf) __attribute__((always_inline)) {
#pragma unroll
    for (int u = 0; u < 4; ++u) {
      int c = tid + u * 128;                 // chunk id 0..511 (16B chunks)
      // K tile: 8 KB contiguous in global
      async_ld_b128(&sK[buf][c * 8], gK + (size_t)kv0 * 64 + c * 8);
      // V^T tile: 64 rows of 128 B, row stride L
      int row = c >> 3, ch = c & 7;
      async_ld_b128(&sV[buf][c * 8], gV + (size_t)row * L + kv0 + ch * 8);
    }
  };

  stage(0, 0);

  const int NIT = L / 64;
  for (int it = 0; it < NIT; ++it) {
    const int cur = it & 1;
    if (it + 1 < NIT) {
      stage((it + 1) * 64, cur ^ 1);
      // current tile's 4 K/V pairs (8 loads) complete (async loads retire in order per wave)
      asm volatile("s_wait_asynccnt 0x8" ::: "memory");
    } else {
      asm volatile("s_wait_asynccnt 0x0" ::: "memory");
    }
    __syncthreads();  // publish tile `cur` block-wide

    // S = Q*K^T for 64 kv columns: 4 C tiles x 2 k-steps
    v8f s[4];
#pragma unroll
    for (int jt = 0; jt < 4; ++jt) {
      v8f acc = {};
#pragma unroll
      for (int st = 0; st < 2; ++st) {
        const bf16_t* pk = &sK[cur][(jt * 16 + lmod) * 64 + st * 32 + lhi * 8];
        v16bf bkf = load16(pk, pk + 16);
        acc = wmma_bf16(aq[st], bkf, acc);
      }
      s[jt] = acc;
    }

    // online softmax (row r+8*lhi lives in this lane's 16-lane half)
#pragma unroll
    for (int r = 0; r < 8; ++r) {
      float sv0 = s[0][r] * scale, sv1 = s[1][r] * scale;
      float sv2 = s[2][r] * scale, sv3 = s[3][r] * scale;
      float rm = fmaxf(fmaxf(sv0, sv1), fmaxf(sv2, sv3));
      rm = fmaxf(rm, __shfl_xor(rm, 1, 32));
      rm = fmaxf(rm, __shfl_xor(rm, 2, 32));
      rm = fmaxf(rm, __shfl_xor(rm, 4, 32));
      rm = fmaxf(rm, __shfl_xor(rm, 8, 32));
      float nm = fmaxf(mrow[r], rm);
      float corr = __expf(mrow[r] - nm);
      mrow[r] = nm;
      float p0 = __expf(sv0 - nm), p1 = __expf(sv1 - nm);
      float p2 = __expf(sv2 - nm), p3 = __expf(sv3 - nm);
      float ps = (p0 + p1) + (p2 + p3);
      ps += __shfl_xor(ps, 1, 32);
      ps += __shfl_xor(ps, 2, 32);
      ps += __shfl_xor(ps, 4, 32);
      ps += __shfl_xor(ps, 8, 32);
      lrow[r] = lrow[r] * corr + ps;
#pragma unroll
      for (int nt = 0; nt < 4; ++nt) o[nt][r] *= corr;
      // C-layout -> row-major 16x64 P tile (wave-private, DS ops are in-order)
      bf16_t* pr = &sP[wid][(r + 8 * lhi) * 64 + lmod];
      pr[0]  = (bf16_t)p0;
      pr[16] = (bf16_t)p1;
      pr[32] = (bf16_t)p2;
      pr[48] = (bf16_t)p3;
    }

    // reload P as two A-fragments (16x32 each) -- same wave, no barrier needed
    v16bf ap[2];
#pragma unroll
    for (int st = 0; st < 2; ++st) {
      const bf16_t* pp = &sP[wid][lmod * 64 + st * 32 + lhi * 8];
      ap[st] = load16(pp, pp + 16);
    }

    // O += P * V
#pragma unroll
    for (int nt = 0; nt < 4; ++nt) {
#pragma unroll
      for (int st = 0; st < 2; ++st) {
        const bf16_t* pv = &sV[cur][(nt * 16 + lmod) * 64 + st * 32 + lhi * 8];
        v16bf bvf = load16(pv, pv + 16);
        o[nt] = wmma_bf16(ap[st], bvf, o[nt]);
      }
    }
    __syncthreads();  // all waves done with tile `cur` before it is re-staged
  }

  // normalize and store to (B, L, D) bf16
  const int b = bh >> 4, h = bh & 15;
#pragma unroll
  for (int nt = 0; nt < 4; ++nt)
#pragma unroll
    for (int r = 0; r < 8; ++r) {
      int row = q0 + r + 8 * lhi;
      int col = h * 64 + nt * 16 + lmod;
      float invl = 1.0f / lrow[r];
      out[((size_t)b * L + row) * 1024 + col] = (bf16_t)(o[nt][r] * invl);
    }
}

// ---------------------------------------------------------------------------
// launch
// ---------------------------------------------------------------------------
extern "C" void kernel_launch(void* const* d_in, const int* in_sizes, int n_in,
                              void* d_out, int out_size, void* d_ws, size_t ws_size,
                              hipStream_t stream) {
  (void)in_sizes; (void)n_in; (void)out_size; (void)ws_size;
  const float* x     = (const float*)d_in[0];
  const float* w_qkv = (const float*)d_in[1];
  const float* w_out = (const float*)d_in[2];
  float* out = (float*)d_out;
  char* ws = (char*)d_ws;

  const size_t B = 2, L = 2048, D = 1024, H = 16;
  const size_t M = B * L;             // 4096
  const size_t nx = B * L * D;        // 4,194,304
  const size_t nwqkv = 3 * D * D;     // 3,145,728
  const size_t nwout = D * D;         // 1,048,576

  // workspace layout (bytes)
  bf16_t* x_bf    = (bf16_t*)(ws);                        //  8 MiB
  bf16_t* wqkv_bf = (bf16_t*)(ws + 8388608);              //  6 MiB
  bf16_t* wout_bf = (bf16_t*)(ws + 14680064);             //  2 MiB
  float*  qkv     = (float*) (ws + 16777216);             // 48 MiB
  bf16_t* qb      = (bf16_t*)(ws + 67108864);             //  8 MiB
  bf16_t* kbuf    = (bf16_t*)(ws + 75497472);             //  8 MiB
  bf16_t* vt      = (bf16_t*)(ws + 83886080);             //  8 MiB
  bf16_t* attn    = (bf16_t*)(ws + 92274688);             //  8 MiB

  f32_to_bf16_kernel<<<(unsigned)((nx + 255) / 256), 256, 0, stream>>>(x, x_bf, nx);
  f32_to_bf16_kernel<<<(unsigned)((nwqkv + 255) / 256), 256, 0, stream>>>(w_qkv, wqkv_bf, nwqkv);
  f32_to_bf16_kernel<<<(unsigned)((nwout + 255) / 256), 256, 0, stream>>>(w_out, wout_bf, nwout);

  // qkv = x @ w_qkv^T : M=4096, N=3072, K=1024
  gemm_bf16_wmma<<<dim3(3072 / 128, (unsigned)(M / 128)), 128, 0, stream>>>(
      x_bf, wqkv_bf, qkv, (int)M, 3072, 1024);

  const size_t nqkv = B * L * 3 * D;
  rope_split_kernel<<<(unsigned)((nqkv + 255) / 256), 256, 0, stream>>>(qkv, qb, kbuf, vt);

  // flash attention: grid (L/64, B*H)
  attn_kernel<<<dim3((unsigned)(L / 64), (unsigned)(B * H)), 128, 0, stream>>>(
      qb, kbuf, vt, attn);

  // out = attn @ w_out^T : M=4096, N=1024, K=1024
  gemm_bf16_wmma<<<dim3(1024 / 128, (unsigned)(M / 128)), 128, 0, stream>>>(
      attn, wout_bf, out, (int)M, 1024, 1024);
}
